// GlobalAttention_309237645858
// MI455X (gfx1250) — matrix operand order repaired and checked by reading, not compiled, not run on hardware
//
#include <hip/hip_runtime.h>
#include <hip/hip_bf16.h>

// ---------------------------------------------------------------------------
// Global (Luong) attention for MI455X / gfx1250, wave32 + WMMA bf16 +
// async global->LDS copies (ASYNCcnt) with double-buffered LDS tiles.
//
//   cvt : X, ctx, W  fp32 -> bf16 workspace (one pass)
//   h   = X @ W^T                  (bf16 WMMA, f32 accum -> bf16 ws)
//   sc  = h @ ctx^T                (bf16 WMMA -> f32 align[t][b][s] in d_out)
//   p   = softmax(sc)              (in-place on align, bf16 P -> ws, aliases Xbf)
//   out = p @ ctx                  (bf16 WMMA -> f32 attn[t][b][d] in d_out)
// ---------------------------------------------------------------------------

typedef __bf16 bf16_t;
typedef __attribute__((ext_vector_type(16))) __bf16 v16bf;
typedef __attribute__((ext_vector_type(8)))  __bf16 v8bf;
typedef __attribute__((ext_vector_type(8)))  float  v8f;
typedef int v4i __attribute__((vector_size(16)));   // matches builtin's V4i param

// Problem shape (fixed by reference setup_inputs)
constexpr int Bn = 16;
constexpr int Tn = 1024;
constexpr int Sn = 1024;
constexpr int Dn = 1024;

constexpr int BLK_M = 128;
constexpr int BLK_N = 128;
constexpr int BLK_K = 32;
constexpr int KT    = Dn / BLK_K;        // 32 K-tiles (Sn == Dn)
constexpr int LDS_STRIDE = BLK_K + 8;    // pad rows (80B) against bank conflicts

#define GLOBAL_AS __attribute__((address_space(1)))
#define LDS_AS    __attribute__((address_space(3)))

// ---- async global->LDS 16B copy (ASYNCcnt-tracked, no VGPR data path) ----
__device__ inline void cp_async_b128(const bf16_t* g, bf16_t* l) {
#if __has_builtin(__builtin_amdgcn_global_load_async_to_lds_b128)
  __builtin_amdgcn_global_load_async_to_lds_b128(
      (GLOBAL_AS v4i*)g, (LDS_AS v4i*)l, /*offset=*/0, /*cpol=*/0);
#else
  asm volatile("global_load_async_to_lds_b128 %0, %1, off"
               :: "v"((unsigned)(unsigned long long)l), "v"(g)
               : "memory");
#endif
}

template <int N>
__device__ inline void wait_async() {
#if __has_builtin(__builtin_amdgcn_s_wait_asynccnt)
  __builtin_amdgcn_s_wait_asynccnt(N);
#else
  if constexpr (N == 0)      asm volatile("s_wait_asynccnt 0" ::: "memory");
  else if constexpr (N == 2) asm volatile("s_wait_asynccnt 2" ::: "memory");
  else                       asm volatile("s_wait_asynccnt 4" ::: "memory");
#endif
}

union V16U { v16bf v; v8bf h[2]; };

// A fragment, 16x32 bf16 (ISA 7.12.2): lane L holds row M = L%16,
// K = (L/16)*8 .. +7 in v[0..3] and K = (L/16)*8+16 .. +7 in v[4..7].
__device__ inline v16bf load_frag_A(const bf16_t (*As)[LDS_STRIDE], int rowBase, int lane) {
  const int lr = lane & 15;
  const int kb = (lane >> 4) * 8;
  V16U u;
  u.h[0] = *(const v8bf*)&As[rowBase + lr][kb];
  u.h[1] = *(const v8bf*)&As[rowBase + lr][kb + 16];
  return u.v;
}

// B fragment, 32x16 bf16: lane L holds column N = L%16,
// K = (L/16)*8 .. +7 in v[0..3] and K = (L/16)*8+16 .. +7 in v[4..7]. Bs is stored [n][k].
__device__ inline v16bf load_frag_B(const bf16_t (*Bs)[LDS_STRIDE], int colBase, int lane) {
  const int lr = lane & 15;
  const int kb = (lane >> 4) * 8;
  V16U u;
  u.h[0] = *(const v8bf*)&Bs[colBase + lr][kb];
  u.h[1] = *(const v8bf*)&Bs[colBase + lr][kb + 16];
  return u.v;
}

// fp32 -> bf16 bulk conversion (8-wide), grid-stride.
__global__ __launch_bounds__(256)
void cvt_f32_to_bf16(const float* __restrict__ src, bf16_t* __restrict__ dst, int n8) {
  int i = blockIdx.x * blockDim.x + threadIdx.x;
  const int stride = gridDim.x * blockDim.x;
  for (; i < n8; i += stride) {
    v8f f = *(const v8f*)(src + (size_t)i * 8);
    *(v8bf*)(dst + (size_t)i * 8) = __builtin_convertvector(f, v8bf);
  }
}

// MODE 0: C[b,t,e] = sum_d X[b,t,d]  * W[e,d]      -> bf16 H ws      (B = [N,K])
// MODE 1: C[b,t,s] = sum_d H[b,t,d]  * ctx[b,s,d]  -> f32 align out  (B = [N,K])
// MODE 2: C[b,t,d] = sum_s P[b,t,s]  * ctx[b,s,d]  -> f32 attn out   (B = [K,N], transposed stage)
template <int MODE>
__global__ __launch_bounds__(256)
void gemm_wmma(const bf16_t* __restrict__ Abase,
               const bf16_t* __restrict__ Bbase,
               bf16_t* __restrict__ outH,
               float* __restrict__ outF)
{
  __shared__ bf16_t As[2][BLK_M][LDS_STRIDE];
  __shared__ bf16_t Bs[2][BLK_N][LDS_STRIDE];

  const int tid   = threadIdx.x;
  const int lane  = tid & 31;
  const int wave  = tid >> 5;       // 8 wave32 per workgroup
  const int waveM = wave >> 2;      // 0..1  (64 rows each)
  const int waveN = wave & 3;       // 0..3  (32 cols each)
  const int b     = blockIdx.z;
  const int mBase = blockIdx.x * BLK_M;
  const int nBase = blockIdx.y * BLK_N;
  constexpr int K = (MODE == 2) ? Sn : Dn;   // 1024 either way; A lda == K

  const bf16_t* A = Abase + (size_t)b * Tn * K;                      // [T,K]
  const bf16_t* B = (MODE == 0) ? Bbase : (Bbase + (size_t)b * Sn * Dn);

  // Async staging geometry (modes 0/1 + all A): 2 chunks of 16B per thread:
  //   idx = tid + c*256 in [0,512): row = idx>>2 (128), ch = idx&3 (4 x 16B = 64B row)
  const int cpRow0 = (tid + 0)   >> 2, cpCh0 = (tid + 0)   & 3;
  const int cpRow1 = (tid + 256) >> 2, cpCh1 = (tid + 256) & 3;
  // Mode-2 B transpose staging: idx -> srow = idx>>4 (32), seg = idx&15 (16 x 8 cols)
  const int trRow0 = (tid + 0)   >> 4, trSeg0 = (tid + 0)   & 15;
  const int trRow1 = (tid + 256) >> 4, trSeg1 = (tid + 256) & 15;

  auto stage = [&](int buf, int kt) {
    // A tile: 128 x 32 bf16, pure async copy.
    cp_async_b128(A + (size_t)(mBase + cpRow0) * K + kt + cpCh0 * 8, &As[buf][cpRow0][cpCh0 * 8]);
    cp_async_b128(A + (size_t)(mBase + cpRow1) * K + kt + cpCh1 * 8, &As[buf][cpRow1][cpCh1 * 8]);
    if constexpr (MODE != 2) {
      // B rows are N, contiguous in K: async copy, same shape as A.
      cp_async_b128(B + (size_t)(nBase + cpRow0) * Dn + kt + cpCh0 * 8, &Bs[buf][cpRow0][cpCh0 * 8]);
      cp_async_b128(B + (size_t)(nBase + cpRow1) * Dn + kt + cpCh1 * 8, &Bs[buf][cpRow1][cpCh1 * 8]);
    } else {
      // B = ctx[s][d] bf16: coalesced 16B read, transposed LDS scatter -> Bs[n][k].
      v8bf f0 = *(const v8bf*)(B + (size_t)(kt + trRow0) * Dn + nBase + trSeg0 * 8);
      v8bf f1 = *(const v8bf*)(B + (size_t)(kt + trRow1) * Dn + nBase + trSeg1 * 8);
      #pragma unroll
      for (int e = 0; e < 8; ++e) Bs[buf][trSeg0 * 8 + e][trRow0] = f0[e];
      #pragma unroll
      for (int e = 0; e < 8; ++e) Bs[buf][trSeg1 * 8 + e][trRow1] = f1[e];
    }
  };

  v8f acc[4][2];
  const v8f vzero = {0.f, 0.f, 0.f, 0.f, 0.f, 0.f, 0.f, 0.f};
  #pragma unroll
  for (int i = 0; i < 4; ++i)
    #pragma unroll
    for (int j = 0; j < 2; ++j) acc[i][j] = vzero;

  stage(0, 0);   // prologue: tile 0 in flight

  for (int ktIdx = 0; ktIdx < KT; ++ktIdx) {
    const int cur = ktIdx & 1;
    const bool hasNext = (ktIdx + 1) < KT;
    if (hasNext) stage(cur ^ 1, (ktIdx + 1) * BLK_K);   // overlap copy of tile i+1

    // Drain tile i's async copies (tile i+1's stay in flight), then rendezvous.
    if constexpr (MODE != 2) { if (hasNext) wait_async<4>(); else wait_async<0>(); }
    else                     { if (hasNext) wait_async<2>(); else wait_async<0>(); }
    __syncthreads();

    v16bf afrag[4];
    v16bf bfrag[2];
    #pragma unroll
    for (int i = 0; i < 4; ++i) afrag[i] = load_frag_A(As[cur], waveM * 64 + i * 16, lane);
    #pragma unroll
    for (int j = 0; j < 2; ++j) bfrag[j] = load_frag_B(Bs[cur], waveN * 32 + j * 16, lane);

    #pragma unroll
    for (int i = 0; i < 4; ++i)
      #pragma unroll
      for (int j = 0; j < 2; ++j)
        acc[i][j] = __builtin_amdgcn_wmma_f32_16x16x32_bf16(
            /*neg_a=*/false, afrag[i], /*neg_b=*/false, bfrag[j],
            /*c_mod=*/(short)0, acc[i][j], /*reuse_a=*/false, /*reuse_b=*/false);

    __syncthreads();   // all reads of buf `cur` done before it is restaged
  }

  // Epilogue: C frag layout (ISA 7.12.2): vgpr v, lane L -> M = v + 8*(L/16), N = L%16.
  const int mWave = mBase + waveM * 64;
  const int nWave = nBase + waveN * 32;
  const int lr  = lane & 15;
  const int hiM = (lane >> 4) * 8;
  #pragma unroll
  for (int i = 0; i < 4; ++i) {
    #pragma unroll
    for (int j = 0; j < 2; ++j) {
      #pragma unroll
      for (int v = 0; v < 8; ++v) {
        const int m = mWave + i * 16 + hiM + v;
        const int n = nWave + j * 16 + lr;
        const float val = acc[i][j][v];
        if (MODE == 0)      outH[(size_t)b * Tn * Dn + (size_t)m * Dn + n] = (bf16_t)val;
        else if (MODE == 1) outF[((size_t)m * Bn + b) * Sn + n] = val;   // align[t][b][s]
        else                outF[((size_t)m * Bn + b) * Dn + n] = val;   // attn [t][b][d]
      }
    }
  }
}

// In-place softmax over the S axis of align[t][b][s]; also emits bf16 P[b][t][s] to ws.
__global__ __launch_bounds__(256)
void softmax_rows(float* __restrict__ alignOut, bf16_t* __restrict__ Pbf)
{
  const int rowIdx = blockIdx.x;       // t*B + b
  const int t = rowIdx >> 4;
  const int b = rowIdx & 15;
  float*  row  = alignOut + (size_t)rowIdx * Sn;
  bf16_t* prow = Pbf + ((size_t)b * Tn + t) * Sn;

  const int tid  = threadIdx.x;
  const int wid  = tid >> 5;
  const int lane = tid & 31;

  float v[4];
  float m = -3.402823e38f;
  #pragma unroll
  for (int i = 0; i < 4; ++i) { v[i] = row[tid + i * 256]; m = fmaxf(m, v[i]); }
  #pragma unroll
  for (int off = 16; off > 0; off >>= 1) m = fmaxf(m, __shfl_xor(m, off, 32));

  __shared__ float redMax[8];
  __shared__ float redSum[8];
  if (lane == 0) redMax[wid] = m;
  __syncthreads();
  m = redMax[0];
  #pragma unroll
  for (int w = 1; w < 8; ++w) m = fmaxf(m, redMax[w]);

  float s = 0.f;
  #pragma unroll
  for (int i = 0; i < 4; ++i) { v[i] = __expf(v[i] - m); s += v[i]; }
  #pragma unroll
  for (int off = 16; off > 0; off >>= 1) s += __shfl_xor(s, off, 32);
  if (lane == 0) redSum[wid] = s;
  __syncthreads();
  s = 0.f;
  #pragma unroll
  for (int w = 0; w < 8; ++w) s += redSum[w];

  const float inv = 1.0f / s;
  #pragma unroll
  for (int i = 0; i < 4; ++i) {
    const float p = v[i] * inv;
    row[tid + i * 256]  = p;
    prow[tid + i * 256] = (bf16_t)p;
  }
}

extern "C" void kernel_launch(void* const* d_in, const int* in_sizes, int n_in,
                              void* d_out, int out_size, void* d_ws, size_t ws_size,
                              hipStream_t stream) {
  (void)in_sizes; (void)n_in; (void)out_size; (void)ws_size;
  const float* input   = (const float*)d_in[0];   // [B,T,D] fp32
  const float* context = (const float*)d_in[1];   // [B,S,D] fp32
  const float* W_in    = (const float*)d_in[2];   // [D,D]  fp32

  float* out  = (float*)d_out;
  float* attn = out;                              // [T,B,D]
  float* alig = out + (size_t)Tn * Bn * Dn;       // [T,B,S]

  // Workspace layout (bf16, all regions rewritten every launch):
  //   [0      .. 32MB) Xbf [B,T,D]   -- later aliased by Pbf [B,T,S] (X dead after GEMM1)
  //   [32MB   .. 64MB) Cbf [B,S,D]
  //   [64MB   .. 66MB) Wbf [D,D]
  //   [66MB   .. 98MB) Hbf [B,T,D]
  char* ws = (char*)d_ws;
  bf16_t* Xbf = (bf16_t*)(ws);
  bf16_t* Pbf = (bf16_t*)(ws);                                   // aliases Xbf
  bf16_t* Cbf = (bf16_t*)(ws + (size_t)Bn * Tn * Dn * 2);
  bf16_t* Wbf = (bf16_t*)(ws + (size_t)Bn * Tn * Dn * 4);
  bf16_t* Hbf = (bf16_t*)(ws + (size_t)Bn * Tn * Dn * 4 + (size_t)Dn * Dn * 2);

  const dim3 blk(256);
  const dim3 grid(Tn / BLK_M, Dn / BLK_N, Bn);    // M=T, N=D or S (both 1024)

  cvt_f32_to_bf16<<<dim3(2048), blk, 0, stream>>>(input,   Xbf, Bn * Tn * Dn / 8);
  cvt_f32_to_bf16<<<dim3(2048), blk, 0, stream>>>(context, Cbf, Bn * Sn * Dn / 8);
  cvt_f32_to_bf16<<<dim3(512),  blk, 0, stream>>>(W_in,    Wbf, Dn * Dn / 8);

  gemm_wmma<0><<<grid, blk, 0, stream>>>(Xbf, Wbf, Hbf, nullptr);
  gemm_wmma<1><<<grid, blk, 0, stream>>>(Hbf, Cbf, nullptr, alig);
  softmax_rows<<<dim3(Tn * Bn), blk, 0, stream>>>(alig, Pbf);
  gemm_wmma<2><<<grid, blk, 0, stream>>>(Pbf, Cbf, nullptr, attn);
}
